// AxialAttention_7215545057672
// MI455X (gfx1250) — matrix operand order repaired and checked by reading, not compiled, not run on hardware
//
#include <hip/hip_runtime.h>
#include <hip/hip_bf16.h>

#define DEV __device__ __forceinline__

typedef __bf16 bf16_t;
typedef __attribute__((ext_vector_type(16))) __bf16 v16bf;
typedef __attribute__((ext_vector_type(8)))  float  v8f;
typedef __attribute__((ext_vector_type(4)))  unsigned int u32x4;
typedef __attribute__((ext_vector_type(4)))  int i32x4;
typedef __attribute__((ext_vector_type(8)))  int i32x8;

static constexpr int NROWS = 65536;   // 256 sequences * 256 positions (per direction)
static constexpr float LN_EPS = 1e-5f;

#ifndef __has_builtin
#define __has_builtin(x) 0
#endif
#if __has_builtin(__builtin_amdgcn_tensor_load_to_lds)
#define HAVE_TDM 1
#warning "CDNA5 TDM path: __builtin_amdgcn_tensor_load_to_lds ENABLED"
#else
#define HAVE_TDM 0
#warning "CDNA5 TDM path: builtin NOT available, manual staging fallback in use"
#endif

// ---------- scalar helpers ----------
DEV bf16_t f2bf(float f) { return (bf16_t)f; }
DEV float  bf2f(bf16_t b) { return (float)b; }
DEV v8f zero8() { v8f z = {0.f,0.f,0.f,0.f,0.f,0.f,0.f,0.f}; return z; }

DEV v8f wmma_bf16(v16bf a, v16bf b, v8f c) {
  // D = A(16x32 bf16) * B(32x16 bf16) + C(16x16 f32)
  return __builtin_amdgcn_wmma_f32_16x16x32_bf16(false, a, false, b, (short)0, c, false, false);
}

// x is (1, I=256, J=256, D=256). dir0 rows: r=(i<<8)+j, elem = r*256+d (contig).
// dir1 rows: r=(j<<8)+i, elem = ((i*256)+j)*256+d = ((r&255)<<16)+((r>>8)<<8)+d.
DEV int x_index(int r, int d, int dir) {
  if (dir == 0) return (r << 8) + d;
  return ((r & 255) << 16) + ((r >> 8) << 8) + d;
}

// ---------- Tensor Data Mover: 2D bf16 tile, global -> LDS ----------
#if HAVE_TDM
DEV void tdm_load_2d(unsigned lds_byte_addr, const void* gaddr,
                     int elems_per_row, int rows, int row_stride_elems) {
  unsigned long long ga = (unsigned long long)gaddr;
  u32x4 g0;
  g0[0] = 1u;                                    // count=1, user descriptor
  g0[1] = lds_byte_addr;                         // D#.lds_addr
  g0[2] = (unsigned)(ga & 0xFFFFFFFFu);          // global_addr[31:0]
  g0[3] = (unsigned)((ga >> 32) & 0x01FFFFFFu) | (2u << 30);  // addr[56:32] | type=2
  i32x8 g1;
  g1[0] = 1 << 16;                               // data_size = 1 (2 bytes/elem)
  g1[1] = (elems_per_row & 0xFFFF) << 16;        // tensor_dim0[15:0]
  g1[2] = ((unsigned)elems_per_row >> 16) | ((rows & 0xFFFF) << 16);  // dim0 hi | dim1 lo
  g1[3] = ((unsigned)rows >> 16) | ((elems_per_row & 0xFFFF) << 16);  // dim1 hi | tile_dim0
  g1[4] = rows & 0xFFFF;                         // tile_dim1 (tile_dim2 = 0)
  g1[5] = row_stride_elems;                      // tensor_dim0_stride[31:0]
  g1[6] = 0;
  g1[7] = 0;
  i32x4 z4 = {0, 0, 0, 0};
#if __clang_major__ >= 23
  i32x8 z8 = {0, 0, 0, 0, 0, 0, 0, 0};
  __builtin_amdgcn_tensor_load_to_lds(g0, g1, z4, z4, z8, 0);
#else
  __builtin_amdgcn_tensor_load_to_lds(g0, g1, z4, z4, 0);
#endif
}
#endif

// ---------- WMMA fragment loaders (per gfx1250 bf16 layouts) ----------
// A 16x32: lane holds row M=lane%16; K = 16g+e contiguous, g=lane/16.
DEV v16bf load_a_frag(const bf16_t* p, int ld) {
  int lane = threadIdx.x & 31;
  const bf16_t* row = p + (lane & 15) * ld + (lane >> 4) * 16;
  v16bf f;
#pragma unroll
  for (int e = 0; e < 16; ++e) f[e] = row[e];
  return f;
}
// B 32x16 staged K-major per column: p[col*ld + k]; lane col=lane%16, K = 16*(lane/16)+e.
DEV v16bf load_b_frag_t(const bf16_t* p, int ld) {
  int lane = threadIdx.x & 31;
  const bf16_t* src = p + (lane & 15) * ld + (lane >> 4) * 16;
  v16bf f;
#pragma unroll
  for (int e = 0; e < 16; ++e) f[e] = src[e];
  return f;
}

// In-register RoPE on an A fragment (contiguous K): d = dbase + e.
DEV void rope_frag_a(v16bf& f, int t, int dbase,
                     const float* __restrict__ sn, const float* __restrict__ cs) {
  const float* st = sn + t * 64;
  const float* ct = cs + t * 64;
#pragma unroll
  for (int p = 0; p < 16; p += 2) {
    int d = dbase + p;
    float x0 = bf2f(f[p]), x1 = bf2f(f[p + 1]);
    f[p]     = f2bf(x0 * ct[d] - x1 * st[d]);
    f[p + 1] = f2bf(x1 * ct[d + 1] + x0 * st[d + 1]);
  }
}
// In-register RoPE on a B fragment (contiguous K): d = dbase + p.
DEV void rope_frag_b(v16bf& f, int t, int dbase,
                     const float* __restrict__ sn, const float* __restrict__ cs) {
  const float* st = sn + t * 64;
  const float* ct = cs + t * 64;
#pragma unroll
  for (int p = 0; p < 16; p += 2) {
    int d = dbase + p;
    float x0 = bf2f(f[p]), x1 = bf2f(f[p + 1]);
    f[p]     = f2bf(x0 * ct[d] - x1 * st[d]);
    f[p + 1] = f2bf(x1 * ct[d + 1] + x0 * st[d + 1]);
  }
}

// cross-lane reductions within a 16-lane half (C-tile rows live across 16 lanes)
DEV float redmax16(float v) {
#pragma unroll
  for (int off = 8; off > 0; off >>= 1) v = fmaxf(v, __shfl_xor(v, off, 32));
  return v;
}
DEV float redsum16(float v) {
#pragma unroll
  for (int off = 8; off > 0; off >>= 1) v += __shfl_xor(v, off, 32);
  return v;
}

// ---------- elementwise kernels ----------
__global__ void k_copy4(const float4* __restrict__ s, float4* __restrict__ d, int n4) {
  int i = blockIdx.x * blockDim.x + threadIdx.x;
  if (i < n4) d[i] = s[i];
}
__global__ void k_conv(const float* __restrict__ s, bf16_t* __restrict__ d, int n) {
  int i = blockIdx.x * blockDim.x + threadIdx.x;
  if (i < n) d[i] = f2bf(s[i]);
}

// Fold LN gain into weights, LN bias into per-column bias:
//   W'[k][c0+n] = bf16(g[k]*W[k][n]);  cvec[c0+n] = sum_k b[k]*W[k][n]
__global__ void k_prep_w(const float* __restrict__ W, const float* __restrict__ g,
                         const float* __restrict__ b, bf16_t* __restrict__ Wc,
                         float* __restrict__ cvec, int K, int N, int ldc, int c0) {
  int n = blockIdx.x * blockDim.x + threadIdx.x;
  if (n >= N) return;
  float acc = 0.f;
  for (int k = 0; k < K; ++k) {
    float w = W[k * N + n];
    Wc[k * ldc + c0 + n] = f2bf(g[k] * w);
    acc += b[k] * w;
  }
  cvec[c0 + n] = acc;
}

// per-row LN stats + normalize: xn[r*256+d] = bf16((x - mu) * rstd), dense layout
__global__ __launch_bounds__(256) void k_stats_norm(const float* __restrict__ x,
                                                    bf16_t* __restrict__ xn, int dir) {
  __shared__ float red[256];
  int r = blockIdx.x, d = threadIdx.x;
  float v = x[x_index(r, d, dir)];
  red[d] = v;
  __syncthreads();
#pragma unroll
  for (int s2 = 128; s2 > 0; s2 >>= 1) { if (d < s2) red[d] += red[d + s2]; __syncthreads(); }
  float mu = red[0] * (1.f / 256.f);
  __syncthreads();
  float dv = v - mu;
  red[d] = dv * dv;
  __syncthreads();
#pragma unroll
  for (int s2 = 128; s2 > 0; s2 >>= 1) { if (d < s2) red[d] += red[d + s2]; __syncthreads(); }
  float rs = rsqrtf(red[0] * (1.f / 256.f) + LN_EPS);
  xn[r * 256 + d] = f2bf(dv * rs);
}

// ---------- QKV GEMM: (65536 x 256) @ (256 x 768), bf16 WMMA ----------
// grid (512, 3): blockIdx.y selects q / k / v output buffer (256-wide n-tile).
__global__ __launch_bounds__(256) void k_qkv(
    const bf16_t* __restrict__ xn, const bf16_t* __restrict__ Wc,
    const float* __restrict__ cvec,
    bf16_t* __restrict__ qb, bf16_t* __restrict__ kb, bf16_t* __restrict__ vb) {
  __shared__ bf16_t a_lds[128 * 32];   // [row][k]
  __shared__ bf16_t bT[256 * 32];      // [col][k]
  const int r0 = blockIdx.x * 128, ny = blockIdx.y, n0 = ny * 256;
  const int tid = threadIdx.x, lane = tid & 31, wave = tid >> 5;
  const int wm = wave & 3, wn = wave >> 2;
  const int g = lane >> 4, ln = lane & 15;

  v8f acc[2][8];
#pragma unroll
  for (int a = 0; a < 2; ++a)
#pragma unroll
    for (int b = 0; b < 8; ++b) acc[a][b] = zero8();

  for (int k0 = 0; k0 < 256; k0 += 32) {
    __syncthreads();
#if HAVE_TDM
    if (wave == 0)   // A tile: 128 rows x 32 bf16, row stride 256 elems -> TDM
      tdm_load_2d((unsigned)(unsigned long long)a_lds, xn + r0 * 256 + k0, 32, 128, 256);
#else
    {
      int row = tid >> 1, part = (tid & 1) * 16;
      const uint4* s4 = (const uint4*)(xn + (r0 + row) * 256 + k0 + part);
      uint4* d4 = (uint4*)&a_lds[row * 32 + part];
      d4[0] = s4[0];
      d4[1] = s4[1];
    }
#endif
    {   // B tile transposed: bT[col][k], col = tid
      int col = tid;
#pragma unroll
      for (int kk = 0; kk < 32; ++kk)
        bT[col * 32 + kk] = Wc[(k0 + kk) * 768 + n0 + col];
      if (k0 + 32 < 256)
        __builtin_prefetch(&Wc[(k0 + 32 + (tid & 31)) * 768 + n0 + (tid >> 5) * 32], 0, 1);
    }
#if HAVE_TDM
    if (wave == 0) __builtin_amdgcn_s_wait_tensorcnt((short)0);
#endif
    __syncthreads();
    v16bf af0 = load_a_frag(&a_lds[(wm * 32) * 32], 32);
    v16bf af1 = load_a_frag(&a_lds[(wm * 32 + 16) * 32], 32);
#pragma unroll
    for (int tn = 0; tn < 8; ++tn) {
      v16bf bf = load_b_frag_t(&bT[(wn * 128 + tn * 16) * 32], 32);
      acc[0][tn] = wmma_bf16(af0, bf, acc[0][tn]);
      acc[1][tn] = wmma_bf16(af1, bf, acc[1][tn]);
    }
  }
  // epilogue: destination buffer uniform per block
  bf16_t* dst = (ny == 0) ? qb : (ny == 1) ? kb : vb;
#pragma unroll
  for (int tn = 0; tn < 8; ++tn) {
    int colbase = wn * 128 + tn * 16;
    float cv = cvec[n0 + colbase + ln];
#pragma unroll
    for (int tm = 0; tm < 2; ++tm)
#pragma unroll
      for (int v = 0; v < 8; ++v) {
        int row = r0 + wm * 32 + tm * 16 + v + 8 * g;
        dst[row * 256 + colbase + ln] = f2bf(acc[tm][tn][v] + cv);
      }
  }
}

// ---------- flash attention with fused RoPE: one block per (sequence, head) ----------
__global__ __launch_bounds__(256) void k_attn(
    const bf16_t* __restrict__ q, const bf16_t* __restrict__ kbuf,
    const bf16_t* __restrict__ vbuf, bf16_t* __restrict__ obuf,
    const float* __restrict__ sn, const float* __restrict__ cs) {
  __shared__ bf16_t k_ch[64 * 64];       // [j][kd]
  __shared__ bf16_t vT[64 * 64];         // [d][j]
  __shared__ bf16_t p_lds[8 * 32 * 64];  // per-wave P (32 x 64)

  const int s = blockIdx.x, h = blockIdx.y;
  const int tid = threadIdx.x, lane = tid & 31, wave = tid >> 5;
  const int g = lane >> 4, ln = lane & 15;

  const bf16_t* qh = q    + (s * 256) * 256 + h * 64;
  const bf16_t* kh = kbuf + (s * 256) * 256 + h * 64;
  const bf16_t* vh = vbuf + (s * 256) * 256 + h * 64;

  // q fragments, RoPE applied in-register (row t = wave*32 + tm*16 + lane%16)
  v16bf qf[2][2];
#pragma unroll
  for (int tm = 0; tm < 2; ++tm)
#pragma unroll
    for (int ks = 0; ks < 2; ++ks) {
      qf[tm][ks] = load_a_frag(qh + (wave * 32 + tm * 16) * 256 + ks * 32, 256);
      rope_frag_a(qf[tm][ks], wave * 32 + tm * 16 + ln, ks * 32 + 16 * g, sn, cs);
    }

  float mrun[16], lrun[16];
#pragma unroll
  for (int i = 0; i < 16; ++i) { mrun[i] = -1e30f; lrun[i] = 0.f; }
  v8f oacc[2][4];
#pragma unroll
  for (int a = 0; a < 2; ++a)
#pragma unroll
    for (int b = 0; b < 4; ++b) oacc[a][b] = zero8();

  bf16_t* pw = p_lds + wave * 32 * 64;

  for (int jc = 0; jc < 4; ++jc) {
    __syncthreads();
#if HAVE_TDM
    if (wave == 0)   // K chunk: 64 rows x 64 bf16, row stride 256 elems -> TDM
      tdm_load_2d((unsigned)(unsigned long long)k_ch, kh + (jc * 64) * 256, 64, 64, 256);
#else
    {
      int jj = tid >> 2, part = (tid & 3) * 16;
      const uint4* s4 = (const uint4*)(kh + (jc * 64 + jj) * 256 + part);
      uint4* d4 = (uint4*)&k_ch[jj * 64 + part];
      d4[0] = s4[0];
      d4[1] = s4[1];
    }
#endif
    {   // V^T chunk [d][jj], manual (transpose)
      int jv = tid & 63, dpart = (tid >> 6) * 16;
      const bf16_t* srcv = vh + (jc * 64 + jv) * 256 + dpart;
#pragma unroll
      for (int e = 0; e < 16; ++e) vT[(dpart + e) * 64 + jv] = srcv[e];
    }
#if HAVE_TDM
    if (wave == 0) __builtin_amdgcn_s_wait_tensorcnt((short)0);
#endif
    __syncthreads();

    // scores S = q k^T for this 64-column chunk (no 1/sqrt(d) in reference)
    v8f sc[2][4];
#pragma unroll
    for (int a = 0; a < 2; ++a)
#pragma unroll
      for (int b = 0; b < 4; ++b) sc[a][b] = zero8();
#pragma unroll
    for (int tn = 0; tn < 4; ++tn)
#pragma unroll
      for (int ks = 0; ks < 2; ++ks) {
        v16bf bf = load_b_frag_t(&k_ch[tn * 16 * 64 + ks * 32], 64);
        // RoPE on k, fused: this lane holds key position j = jc*64 + tn*16 + ln
        rope_frag_b(bf, jc * 64 + tn * 16 + ln, ks * 32 + 16 * g, sn, cs);
        sc[0][tn] = wmma_bf16(qf[0][ks], bf, sc[0][tn]);
        sc[1][tn] = wmma_bf16(qf[1][ks], bf, sc[1][tn]);
      }

    // online softmax update + stage P (bf16) into per-wave LDS
#pragma unroll
    for (int tm = 0; tm < 2; ++tm)
#pragma unroll
      for (int v = 0; v < 8; ++v) {
        float loc = fmaxf(fmaxf(sc[tm][0][v], sc[tm][1][v]),
                          fmaxf(sc[tm][2][v], sc[tm][3][v]));
        float mx = redmax16(loc);
        int ri = tm * 8 + v;
        float mnew = fmaxf(mrun[ri], mx);
        float alpha = __expf(mrun[ri] - mnew);
        float psum = 0.f;
        int prow = (tm * 16 + v + 8 * g) * 64;
#pragma unroll
        for (int tn = 0; tn < 4; ++tn) {
          float pv = __expf(sc[tm][tn][v] - mnew);
          psum += pv;
          pw[prow + tn * 16 + ln] = f2bf(pv);
        }
        psum = redsum16(psum);
        lrun[ri] = lrun[ri] * alpha + psum;
        mrun[ri] = mnew;
#pragma unroll
        for (int tno = 0; tno < 4; ++tno) oacc[tm][tno][v] *= alpha;
      }

    // O += P @ V  (K = 64 j-positions, two 32-steps)
#pragma unroll
    for (int js = 0; js < 2; ++js) {
      v16bf pa0 = load_a_frag(pw + js * 32, 64);
      v16bf pa1 = load_a_frag(pw + 16 * 64 + js * 32, 64);
#pragma unroll
      for (int tno = 0; tno < 4; ++tno) {
        v16bf bv = load_b_frag_t(&vT[tno * 16 * 64 + js * 32], 64);
        oacc[0][tno] = wmma_bf16(pa0, bv, oacc[0][tno]);
        oacc[1][tno] = wmma_bf16(pa1, bv, oacc[1][tno]);
      }
    }
  }

  // write head-interleaved output: feature = d*HEADS + h
#pragma unroll
  for (int tm = 0; tm < 2; ++tm)
#pragma unroll
    for (int v = 0; v < 8; ++v) {
      int t = wave * 32 + tm * 16 + v + 8 * g;
      float inv = 1.f / lrun[tm * 8 + v];
#pragma unroll
      for (int tno = 0; tno < 4; ++tno) {
        int d = tno * 16 + ln;
        obuf[(s * 256 + t) * 256 + d * 4 + h] = f2bf(oacc[tm][tno][v] * inv);
      }
    }
}

// ---------- output projection + ELU + residual accumulate ----------
__global__ __launch_bounds__(256) void k_proj(
    const bf16_t* __restrict__ ob, const bf16_t* __restrict__ Woc,
    const float* __restrict__ bo, float* __restrict__ out, int dir) {
  __shared__ bf16_t a_lds[128 * 32];
  __shared__ bf16_t bT[64 * 32];
  const int r0 = blockIdx.x * 128, n0 = blockIdx.y * 64;
  const int tid = threadIdx.x, lane = tid & 31, wave = tid >> 5;
  const int wm = wave & 3, wn = wave >> 2;
  const int g = lane >> 4, ln = lane & 15;

  v8f acc[2][2];
#pragma unroll
  for (int a = 0; a < 2; ++a)
#pragma unroll
    for (int b = 0; b < 2; ++b) acc[a][b] = zero8();

  for (int k0 = 0; k0 < 256; k0 += 32) {
    __syncthreads();
#if HAVE_TDM
    if (wave == 0)   // A tile: 128 rows x 32 bf16, row stride 256 elems -> TDM
      tdm_load_2d((unsigned)(unsigned long long)a_lds, ob + r0 * 256 + k0, 32, 128, 256);
#else
    {
      int row = tid >> 1, part = (tid & 1) * 16;
      const uint4* s4 = (const uint4*)(ob + (r0 + row) * 256 + k0 + part);
      uint4* d4 = (uint4*)&a_lds[row * 32 + part];
      d4[0] = s4[0];
      d4[1] = s4[1];
    }
#endif
    {
      int col = tid >> 2, kp = (tid & 3) * 8;
#pragma unroll
      for (int e = 0; e < 8; ++e)
        bT[col * 32 + kp + e] = Woc[(k0 + kp + e) * 256 + n0 + col];
      if (k0 + 32 < 256)
        __builtin_prefetch(&Woc[(k0 + 32 + kp) * 256 + n0 + col], 0, 1);
    }
#if HAVE_TDM
    if (wave == 0) __builtin_amdgcn_s_wait_tensorcnt((short)0);
#endif
    __syncthreads();
    v16bf af0 = load_a_frag(&a_lds[(wm * 32) * 32], 32);
    v16bf af1 = load_a_frag(&a_lds[(wm * 32 + 16) * 32], 32);
#pragma unroll
    for (int tn = 0; tn < 2; ++tn) {
      v16bf bf = load_b_frag_t(&bT[(wn * 32 + tn * 16) * 32], 32);
      acc[0][tn] = wmma_bf16(af0, bf, acc[0][tn]);
      acc[1][tn] = wmma_bf16(af1, bf, acc[1][tn]);
    }
  }
#pragma unroll
  for (int tm = 0; tm < 2; ++tm)
#pragma unroll
    for (int tn = 0; tn < 2; ++tn)
#pragma unroll
      for (int v = 0; v < 8; ++v) {
        int row = r0 + wm * 32 + tm * 16 + v + 8 * g;
        int col = n0 + wn * 32 + tn * 16 + ln;
        float y = acc[tm][tn][v] + bo[col];
        y = y > 0.f ? y : (__expf(y) - 1.f);            // ELU
        out[x_index(row, col, dir)] += 0.5f * y;        // x + (xi + xj)/2
      }
}

// ---------- host orchestration ----------
extern "C" void kernel_launch(void* const* d_in, const int* in_sizes, int n_in,
                              void* d_out, int out_size, void* d_ws, size_t ws_size,
                              hipStream_t stream) {
  (void)in_sizes; (void)n_in; (void)out_size; (void)ws_size;
  const float* x     = (const float*)d_in[0];
  const float* sin_i = (const float*)d_in[1];
  const float* cos_i = (const float*)d_in[2];
  const float* sin_j = (const float*)d_in[3];
  const float* cos_j = (const float*)d_in[4];
  float* out = (float*)d_out;

  char* p = (char*)d_ws;
  auto carve = [&](size_t bytes) -> void* {
    void* r = (void*)p;
    p += (bytes + 255) & ~(size_t)255;
    return r;
  };
  bf16_t* xn    = (bf16_t*)carve((size_t)NROWS * 256 * 2);
  bf16_t* qb    = (bf16_t*)carve((size_t)NROWS * 256 * 2);
  bf16_t* kb    = (bf16_t*)carve((size_t)NROWS * 256 * 2);
  bf16_t* vb    = (bf16_t*)carve((size_t)NROWS * 256 * 2);
  bf16_t* ob    = (bf16_t*)carve((size_t)NROWS * 256 * 2);
  bf16_t* Wc    = (bf16_t*)carve((size_t)256 * 768 * 2);
  float*  cvec  = (float*)carve(768 * 4);
  bf16_t* Woc   = (bf16_t*)carve((size_t)256 * 256 * 2);

  k_copy4<<<NROWS / 4, 256, 0, stream>>>((const float4*)x, (float4*)out, NROWS * 64);

  for (int dir = 0; dir < 2; ++dir) {
    const float* sn = dir ? sin_j : sin_i;
    const float* cs = dir ? cos_j : cos_i;
    const float* ga  = (const float*)d_in[dir ? 13 : 5];
    const float* ba  = (const float*)d_in[dir ? 14 : 6];
    const float* gb  = (const float*)d_in[dir ? 15 : 7];
    const float* bb  = (const float*)d_in[dir ? 16 : 8];
    const float* Wq  = (const float*)d_in[dir ? 17 : 9];
    const float* Wkv = (const float*)d_in[dir ? 18 : 10];
    const float* Wo  = (const float*)d_in[dir ? 19 : 11];
    const float* bo  = (const float*)d_in[dir ? 20 : 12];

    k_prep_w<<<1, 256, 0, stream>>>(Wq, ga, ba, Wc, cvec, 256, 256, 768, 0);
    k_prep_w<<<2, 256, 0, stream>>>(Wkv, gb, bb, Wc, cvec, 256, 512, 768, 256);
    k_conv<<<256, 256, 0, stream>>>(Wo, Woc, 256 * 256);

    // LayerNorm (gain/bias folded into weights) -> dense bf16 x_hat
    k_stats_norm<<<NROWS, 256, 0, stream>>>(x, xn, dir);

    dim3 gq(NROWS / 128, 3);
    k_qkv<<<gq, 256, 0, stream>>>(xn, Wc, cvec, qb, kb, vb);

    dim3 ga2(256, 4);
    k_attn<<<ga2, 256, 0, stream>>>(qb, kb, vb, ob, sn, cs);

    dim3 gp(NROWS / 128, 4);
    k_proj<<<gp, 256, 0, stream>>>(ob, Woc, bo, out, dir);
  }
}
